// DeepEmbCluster_49331994362285
// MI455X (gfx1250) — matrix operand order repaired and checked by reading, not compiled, not run on hardware
//
#include <hip/hip_runtime.h>

// ---------------------------------------------------------------------------
// DEC soft-assignment (Student-t) for MI455X / gfx1250.
//   cross = emb @ centers^T   via v_wmma_f32_16x16x32_bf16 (fp32 accum)
//   q     = (1 + d2/alpha)^(-(alpha+1)/2), row-normalized, fully fused.
// emb: [65536 x 768] f32, centers: [512 x 768] f32, out: [65536 x 512] f32.
//
// Pipelining: all 8 B fragments of a k-chunk are preloaded into registers
// (single load clause, overlapped waits), and the LDS A stage is double-
// buffered so the next chunk's global loads + ds_stores run under the
// current chunk's 16 WMMAs, with one barrier per chunk.
// ---------------------------------------------------------------------------

typedef __attribute__((ext_vector_type(16))) __bf16 v16bf;
typedef __attribute__((ext_vector_type(8)))  __bf16 v8bf;
typedef __attribute__((ext_vector_type(8)))  float  v8f;
typedef __attribute__((ext_vector_type(4)))  float  v4f;

#define N_EMB 65536
#define K_CL  512
#define D_DIM 768
#define BM    64          // output rows per workgroup (x full 512 cols)
#define SLDA  48          // LDS row stride in bf16 (16B-aligned, bank-spread)

// ---------------- prep: centers f32 -> bf16 (+ per-cluster ||c||^2) --------
__global__ void dec_prep_centers(const float* __restrict__ centers,
                                 __bf16* __restrict__ cbf,
                                 float* __restrict__ csq) {
  __shared__ float red[256];
  const int k = blockIdx.x;          // cluster index
  const int t = threadIdx.x;
  const float* row = centers + (size_t)k * D_DIM;
  __bf16* orow = cbf + (size_t)k * D_DIM;
  float ss = 0.0f;
  for (int i = t; i < D_DIM; i += 256) {
    float v = row[i];
    ss += v * v;
    orow[i] = (__bf16)v;
  }
  red[t] = ss;
  __syncthreads();
  for (int s = 128; s > 0; s >>= 1) {
    if (t < s) red[t] += red[t + s];
    __syncthreads();
  }
  if (t == 0) csq[k] = red[0];
}

// ---------------- main fused GEMM + Student-t + row-normalize --------------
__global__ __launch_bounds__(256, 1)
void dec_main(const float* __restrict__ emb,
              const __bf16* __restrict__ cbf,
              const float* __restrict__ csq,
              const float* __restrict__ alpha_p,
              float* __restrict__ out) {
  __shared__ __bf16 s_a[2][BM * SLDA];  // double-buffered A tile (bf16)
  __shared__ float  s_xsq[BM];          // per-row ||x||^2
  __shared__ float  s_rs[BM];           // per-row numerator sum

  const int tid  = threadIdx.x;
  const int lane = tid & 31;
  const int wave = tid >> 5;          // 0..7
  const int wm   = wave >> 2;         // 0..1  (32-row half)
  const int wn   = wave & 3;          // 0..3  (128-col quarter)
  const int lhi  = lane >> 4;         // 0/1
  const int llo  = lane & 15;

  const int row0 = blockIdx.x * BM;

  if (tid < BM) { s_xsq[tid] = 0.0f; s_rs[tid] = 0.0f; }

  // 2 M-tiles x 8 N-tiles of 16x16 f32 accumulators per wave
  v8f acc[2][8];
  #pragma unroll
  for (int m = 0; m < 2; ++m)
    #pragma unroll
    for (int t = 0; t < 8; ++t)
      acc[m][t] = v8f{};

  // A staging assignment: thread -> (row, 8-col strip), fully coalesced
  const int sr = tid >> 2;            // 0..63
  const int sc = (tid & 3) * 8;       // 0,8,16,24
  const float* gA = emb + (size_t)(row0 + sr) * D_DIM + sc;

  // ---- prologue: stage chunk 0 into buffer 0 ----
  {
    v4f a0 = *reinterpret_cast<const v4f*>(gA);
    v4f a1 = *reinterpret_cast<const v4f*>(gA + 4);
    v8bf ab;
    #pragma unroll
    for (int i = 0; i < 4; ++i) {
      ab[i]     = (__bf16)a0[i];
      ab[4 + i] = (__bf16)a1[i];
    }
    *reinterpret_cast<v8bf*>(&s_a[0][sr * SLDA + sc]) = ab;
    float ss = a0[0]*a0[0] + a0[1]*a0[1] + a0[2]*a0[2] + a0[3]*a0[3]
             + a1[0]*a1[0] + a1[1]*a1[1] + a1[2]*a1[2] + a1[3]*a1[3];
    atomicAdd(&s_xsq[sr], ss);
  }
  __syncthreads();

  int buf = 0;
  for (int kb = 0; kb < D_DIM; kb += 32, buf ^= 1) {
    const bool has_next = (kb + 32 < D_DIM);

    // ---- issue next A chunk's global loads early (latency overlap) ----
    v4f na0 = v4f{}, na1 = v4f{};
    if (has_next) {
      na0 = *reinterpret_cast<const v4f*>(gA + kb + 32);
      na1 = *reinterpret_cast<const v4f*>(gA + kb + 36);
      if (kb + 64 < D_DIM) __builtin_prefetch(gA + kb + 64, 0, 1);
    }

    // ---- preload ALL 8 B fragments for this chunk (one load clause) ----
    // B 32x16 layout: lane -> col N=llo; K = lhi*8 + {0..7} in v[0..7], +16 above.
    v16bf bfrag[8];
    #pragma unroll
    for (int t = 0; t < 8; ++t) {
      const __bf16* bp = cbf + (size_t)(wn * 128 + t * 16 + llo) * D_DIM
                             + kb + lhi * 8;
      v8bf b0 = *reinterpret_cast<const v8bf*>(bp);
      v8bf b1 = *reinterpret_cast<const v8bf*>(bp + 16);
      bfrag[t] = __builtin_shufflevector(b0, b1,
                   0,1,2,3,4,5,6,7,8,9,10,11,12,13,14,15);
    }

    // ---- A fragments from current LDS buffer (ISA 16-bit A 16x32 layout:
    //      lane = row M = llo; K = lhi*8 + {0..7} in v[0..7], +16 above) ----
    v16bf afrag[2];
    #pragma unroll
    for (int m = 0; m < 2; ++m) {
      const int base = (wm * 32 + m * 16 + llo) * SLDA + lhi * 8;
      v8bf h0 = *reinterpret_cast<const v8bf*>(&s_a[buf][base]);
      v8bf h1 = *reinterpret_cast<const v8bf*>(&s_a[buf][base + 16]);
      afrag[m] = __builtin_shufflevector(h0, h1,
                   0,1,2,3,4,5,6,7,8,9,10,11,12,13,14,15);
    }

    // ---- stage next chunk into the other buffer while WMMAs run ----
    if (has_next) {
      v8bf ab;
      #pragma unroll
      for (int i = 0; i < 4; ++i) {
        ab[i]     = (__bf16)na0[i];
        ab[4 + i] = (__bf16)na1[i];
      }
      *reinterpret_cast<v8bf*>(&s_a[buf ^ 1][sr * SLDA + sc]) = ab;
      float ss = na0[0]*na0[0] + na0[1]*na0[1] + na0[2]*na0[2] + na0[3]*na0[3]
               + na1[0]*na1[0] + na1[1]*na1[1] + na1[2]*na1[2] + na1[3]*na1[3];
      atomicAdd(&s_xsq[sr], ss);
    }

    // ---- 16 WMMAs; B loads drain in order as t advances ----
    #pragma unroll
    for (int m = 0; m < 2; ++m)
      #pragma unroll
      for (int t = 0; t < 8; ++t)
        acc[m][t] = __builtin_amdgcn_wmma_f32_16x16x32_bf16(
            false, afrag[m], false, bfrag[t], (short)0, acc[m][t],
            false, false);

    __syncthreads();   // next-buffer stores visible; current buffer reusable
  }

  // ---------------- fused epilogue ----------------
  const float alpha = alpha_p[0];
  const float inv_a = 1.0f / alpha;
  const float pexp  = -(alpha + 1.0f) * 0.5f;

  float xr[2][8];
  #pragma unroll
  for (int m = 0; m < 2; ++m)
    #pragma unroll
    for (int v = 0; v < 8; ++v)
      xr[m][v] = s_xsq[wm * 32 + m * 16 + v + 8 * lhi];

  float cs[8];
  #pragma unroll
  for (int t = 0; t < 8; ++t)
    cs[t] = csq[wn * 128 + t * 16 + llo];

  // numerator = (1 + d2/alpha)^(-(alpha+1)/2), d2 = ||x||^2 - 2x.c + ||c||^2
  #pragma unroll
  for (int m = 0; m < 2; ++m)
    #pragma unroll
    for (int t = 0; t < 8; ++t)
      #pragma unroll
      for (int v = 0; v < 8; ++v) {
        float d2 = fmaxf(xr[m][v] + cs[t] - 2.0f * acc[m][t][v], 0.0f);
        acc[m][t][v] = __expf(pexp * __logf(1.0f + d2 * inv_a));
      }

  // cross-wave row sums over all 512 columns
  #pragma unroll
  for (int m = 0; m < 2; ++m)
    #pragma unroll
    for (int v = 0; v < 8; ++v) {
      float p = 0.0f;
      #pragma unroll
      for (int t = 0; t < 8; ++t) p += acc[m][t][v];
      atomicAdd(&s_rs[wm * 32 + m * 16 + v + 8 * lhi], p);
    }
  __syncthreads();

  // normalize + store (lanes 0-15 / 16-31 each write a 64B-contiguous run)
  #pragma unroll
  for (int m = 0; m < 2; ++m)
    #pragma unroll
    for (int v = 0; v < 8; ++v) {
      const int row_l = wm * 32 + m * 16 + v + 8 * lhi;
      const float inv_rs = 1.0f / s_rs[row_l];
      const size_t ob = (size_t)(row0 + row_l) * K_CL + wn * 128 + llo;
      #pragma unroll
      for (int t = 0; t < 8; ++t)
        out[ob + t * 16] = acc[m][t][v] * inv_rs;
    }
}

// ---------------------------------------------------------------------------
extern "C" void kernel_launch(void* const* d_in, const int* in_sizes, int n_in,
                              void* d_out, int out_size, void* d_ws, size_t ws_size,
                              hipStream_t stream) {
  const float* emb     = (const float*)d_in[0];   // [65536 x 768] f32
  const float* centers = (const float*)d_in[1];   // [512 x 768] f32
  const float* alpha   = (const float*)d_in[2];   // scalar f32

  // workspace: bf16 centers (768 KB) then csq (2 KB)
  __bf16* cbf = (__bf16*)d_ws;
  float*  csq = (float*)((char*)d_ws + (size_t)K_CL * D_DIM * sizeof(__bf16));

  dec_prep_centers<<<K_CL, 256, 0, stream>>>(centers, cbf, csq);
  dec_main<<<N_EMB / BM, 256, 0, stream>>>(emb, cbf, csq, alpha, (float*)d_out);
}
